// ParallelAttentionBlock_65687229825659
// MI455X (gfx1250) — matrix operand order repaired and checked by reading, not compiled, not run on hardware
//
#include <hip/hip_runtime.h>
#include <hip/hip_bf16.h>

typedef __attribute__((ext_vector_type(16))) _Float16 v16h;
typedef __attribute__((ext_vector_type(8)))  _Float16 v8h;
typedef __attribute__((ext_vector_type(8)))  float    v8f;
typedef __attribute__((ext_vector_type(4)))  unsigned int v4u;
typedef __attribute__((ext_vector_type(4)))  int      v4i;
typedef __attribute__((ext_vector_type(8)))  int      v8i;

#define D_MODEL  1024
#define N_HEADS  16
#define HEAD_DIM 64
#define N_KV     4
#define KV_DIM   256   // N_KV * HEAD_DIM
#define D_FF     4096
#define SEQ      2048
#define BATCH    2
#define TOKENS   (BATCH * SEQ)   // 4096

// LDS tile row pitch: 64 data + 8 pad halves (16B) -> conflict-free frag reads
#define LPITCH   72
#define TILE_ELE (64 * LPITCH + 16)

#if __has_builtin(__builtin_amdgcn_tensor_load_to_lds)
#define HAS_TDM 1
#else
#define HAS_TDM 0
#endif

// ---------------------------------------------------------------------------
// WMMA helpers (CDNA5 16x16x32 f16 -> f32)
// ---------------------------------------------------------------------------
__device__ __forceinline__ v8f wmma_f16(v16h a, v16h b, v8f c) {
  return __builtin_amdgcn_wmma_f32_16x16x32_f16(
      /*neg_a=*/false, a, /*neg_b=*/false, b,
      /*c_mod=*/(short)0, c, /*reuse_a=*/false, /*reuse_b=*/false);
}

// A-fragment (16x32, 16-bit): lane row m = lane&15.
// k = half*16 + e (contiguous). Two 16B loads.
__device__ __forceinline__ v16h load_a_frag(const _Float16* p, int half) {
  v8h lo = *(const v8h*)(p + half * 16);
  v8h hi = *(const v8h*)(p + half * 16 + 8);
  v16h a;
#pragma unroll
  for (int e = 0; e < 8; ++e) { a[e] = lo[e]; a[8 + e] = hi[e]; }
  return a;
}

// ---------------------------------------------------------------------------
// Tensor Data Mover: 2D tile load (rows x 64 f16) global -> LDS, with LDS
// row padding of 16B per 128B row (pad_interval=32 DW, pad_amount=4 DW) so
// the staged tile lands with a 72-element pitch.
// D# layout per CDNA5 ISA ch.8 (group0 128b, group1 256b).
// ---------------------------------------------------------------------------
#if HAS_TDM
__device__ __forceinline__ void tdm_load_2d(const _Float16* gsrc,
                                            unsigned lds_byte_addr,
                                            int rows, int row_stride_elems) {
  unsigned long long ga = (unsigned long long)(uintptr_t)gsrc;
  v4u g0;
  g0.x = 1u;                                        // count=1 valid descriptor
  g0.y = lds_byte_addr;                             // LDS dest (bytes)
  g0.z = (unsigned)(ga & 0xFFFFFFFFu);              // global_addr[31:0]
  g0.w = (unsigned)((ga >> 32) & 0x1FFFFFFu) | (2u << 30);  // [56:32] | type=2

  unsigned td0 = 64u, td1 = (unsigned)rows;         // tensor dims (= tile dims)
  unsigned tl0 = 64u, tl1 = (unsigned)rows;         // tile dims
  unsigned long long s0 = (unsigned long long)row_stride_elems;
  v8i g1;
  g1[0] = (int)((1u << 16)        // data_size = 1 -> 2 bytes
              | (1u << 20)        // pad_enable
              | (4u << 22)        // pad_interval: 32 DWORDs (=128B row)
              | (3u << 25));      // pad_amount:   4 DWORDs  (=16B)
  g1[1] = (int)((td0 & 0xFFFFu) << 16);                       // tdim0[15:0]
  g1[2] = (int)(((td0 >> 16) & 0xFFFFu) | ((td1 & 0xFFFFu) << 16));
  g1[3] = (int)(((td1 >> 16) & 0xFFFFu) | (tl0 << 16));       // tile_dim0
  g1[4] = (int)(tl1);                                         // tile_dim1, tile_dim2=0
  g1[5] = (int)(s0 & 0xFFFFFFFFull);                          // dim0 stride lo
  g1[6] = (int)((s0 >> 32) & 0xFFFFull);                      // stride hi, stride1=0
  g1[7] = 0;
  v4i gz = {0, 0, 0, 0};
#if defined(__clang_major__) && (__clang_major__ >= 23)
  v8i gz8 = {0, 0, 0, 0, 0, 0, 0, 0};
  __builtin_amdgcn_tensor_load_to_lds(g0, g1, gz, gz, gz8, 0);
#else
  __builtin_amdgcn_tensor_load_to_lds(g0, g1, gz, gz, 0);
#endif
}
#endif

__device__ __forceinline__ unsigned lds_addr_of(const void* p) {
  return (unsigned)(unsigned long long)(uintptr_t)p;   // low 32b = LDS offset
}

// Cooperative fallback copy (no TDM builtin): rows x 64 f16, pitch 72 in LDS
__device__ __forceinline__ void copy_tile(const _Float16* g, _Float16* lds,
                                          int rows, int stride) {
  for (int idx = threadIdx.x; idx < rows * 8; idx += 32) {
    int r = idx >> 3, c = (idx & 7) * 8;
    *(v8h*)(lds + r * LPITCH + c) = *(const v8h*)(g + (size_t)r * stride + c);
  }
}

// ---------------------------------------------------------------------------
// Weight cast + transpose: W[K x N] f32 (row-major) -> Wt[N x K] f16
// ---------------------------------------------------------------------------
__global__ void castTransposeKernel(const float* __restrict__ W,
                                    _Float16* __restrict__ Wt,
                                    int K, int N) {
  long idx = (long)blockIdx.x * 256 + threadIdx.x;
  long total = (long)K * N;
  if (idx < total) {
    int k = (int)(idx / N);
    int n = (int)(idx % N);
    Wt[(size_t)n * K + k] = (_Float16)W[idx];
  }
}

// ---------------------------------------------------------------------------
// LayerNorm over D_MODEL=1024, one row per 256-thread block, f16 output
// ---------------------------------------------------------------------------
__global__ __launch_bounds__(256) void lnKernel(const float* __restrict__ x,
                                                const float* __restrict__ w,
                                                const float* __restrict__ b,
                                                _Float16* __restrict__ out) {
  int row = blockIdx.x;
  int tid = threadIdx.x;
  const float* xr = x + (size_t)row * D_MODEL;
  float v[4], s = 0.f, ss = 0.f;
#pragma unroll
  for (int j = 0; j < 4; ++j) {
    v[j] = xr[tid + j * 256];
    s += v[j];
    ss += v[j] * v[j];
  }
  __shared__ float r0[256], r1[256];
  r0[tid] = s; r1[tid] = ss;
  __syncthreads();
  for (int o = 128; o > 0; o >>= 1) {
    if (tid < o) { r0[tid] += r0[tid + o]; r1[tid] += r1[tid + o]; }
    __syncthreads();
  }
  float mean = r0[0] * (1.0f / D_MODEL);
  float var  = r1[0] * (1.0f / D_MODEL) - mean * mean;
  float inv  = rsqrtf(var + 1e-5f);
#pragma unroll
  for (int j = 0; j < 4; ++j) {
    int c = tid + j * 256;
    out[(size_t)row * D_MODEL + c] = (_Float16)((v[j] - mean) * inv * w[c] + b[c]);
  }
}

// ---------------------------------------------------------------------------
// 64x64 register-blocked multiply-accumulate from staged LDS tiles
// ---------------------------------------------------------------------------
__device__ __forceinline__ void mma_64x64(const _Float16* Al, const _Float16* Bl,
                                          v8f acc[4][4], int half, int ln) {
#pragma unroll
  for (int kk = 0; kk < 64; kk += 32) {
    v16h af[4], bf[4];
#pragma unroll
    for (int ms = 0; ms < 4; ++ms)
      af[ms] = load_a_frag(Al + (ms * 16 + ln) * LPITCH + kk, half);
#pragma unroll
    for (int ns = 0; ns < 4; ++ns)
      bf[ns] = *(const v16h*)(Bl + (ns * 16 + ln) * LPITCH + kk + half * 16);
#pragma unroll
    for (int ms = 0; ms < 4; ++ms)
#pragma unroll
      for (int ns = 0; ns < 4; ++ns)
        acc[ms][ns] = wmma_f16(af[ms], bf[ns], acc[ms][ns]);
  }
}

// ---------------------------------------------------------------------------
// GEMM: C[M x N] = A[M x K] f16 * Bt[N x K] f16. One wave per 64x64 C tile.
// TDM double-buffered LDS staging. mode: 0=f32, 1=f16, 2=f16 transposed.
// ---------------------------------------------------------------------------
__global__ __launch_bounds__(32) void gemmWmmaTiled(const _Float16* __restrict__ A,
                                                    const _Float16* __restrict__ Bt,
                                                    void* __restrict__ Cptr,
                                                    int M, int N, int K,
                                                    int mode, int ldt) {
  __shared__ __align__(16) _Float16 Alds[2][TILE_ELE];
  __shared__ __align__(16) _Float16 Blds[2][TILE_ELE];
  int lane = threadIdx.x, half = lane >> 4, ln = lane & 15;
  int tn = blockIdx.x * 64, tm = blockIdx.y * 64;
  const _Float16* Ag = A  + (size_t)tm * K;
  const _Float16* Bg = Bt + (size_t)tn * K;

  v8f acc[4][4];
#pragma unroll
  for (int ms = 0; ms < 4; ++ms)
#pragma unroll
    for (int ns = 0; ns < 4; ++ns) acc[ms][ns] = (v8f){};

  const int nsteps = K / 64;
#if HAS_TDM
  tdm_load_2d(Ag, lds_addr_of(&Alds[0][0]), 64, K);
  tdm_load_2d(Bg, lds_addr_of(&Blds[0][0]), 64, K);
  for (int s = 0; s < nsteps; ++s) {
    int cur = s & 1;
    if (s + 1 < nsteps) {
      tdm_load_2d(Ag + (s + 1) * 64, lds_addr_of(&Alds[cur ^ 1][0]), 64, K);
      tdm_load_2d(Bg + (s + 1) * 64, lds_addr_of(&Blds[cur ^ 1][0]), 64, K);
      __builtin_amdgcn_s_wait_tensorcnt(2);   // current buffer complete
    } else {
      __builtin_amdgcn_s_wait_tensorcnt(0);
    }
    __syncthreads();
    mma_64x64(&Alds[cur][0], &Blds[cur][0], acc, half, ln);
    __syncthreads();
  }
#else
  for (int s = 0; s < nsteps; ++s) {
    copy_tile(Ag + s * 64, &Alds[0][0], 64, K);
    copy_tile(Bg + s * 64, &Blds[0][0], 64, K);
    __syncthreads();
    mma_64x64(&Alds[0][0], &Blds[0][0], acc, half, ln);
    __syncthreads();
  }
#endif

  // C/D layout: element i -> row m = i + 8*half, col n = lane&15 (per subtile)
  if (mode == 0) {
    float* C = (float*)Cptr;
#pragma unroll
    for (int ms = 0; ms < 4; ++ms)
#pragma unroll
      for (int ns = 0; ns < 4; ++ns)
#pragma unroll
        for (int i = 0; i < 8; ++i)
          C[(size_t)(tm + ms * 16 + i + 8 * half) * N + tn + ns * 16 + ln] = acc[ms][ns][i];
  } else if (mode == 1) {
    _Float16* C = (_Float16*)Cptr;
#pragma unroll
    for (int ms = 0; ms < 4; ++ms)
#pragma unroll
      for (int ns = 0; ns < 4; ++ns)
#pragma unroll
        for (int i = 0; i < 8; ++i)
          C[(size_t)(tm + ms * 16 + i + 8 * half) * N + tn + ns * 16 + ln] =
              (_Float16)acc[ms][ns][i];
  } else {
    _Float16* C = (_Float16*)Cptr;
#pragma unroll
    for (int ms = 0; ms < 4; ++ms)
#pragma unroll
      for (int ns = 0; ns < 4; ++ns)
#pragma unroll
        for (int i = 0; i < 8; ++i)
          C[(size_t)(tn + ns * 16 + ln) * ldt + tm + ms * 16 + i + 8 * half] =
              (_Float16)acc[ms][ns][i];
  }
}

// ---------------------------------------------------------------------------
// Fused SwiGLU gate/up: H = silu(A*Wg) * (A*Wu). Wave tile 64(M) x 32(N).
// ---------------------------------------------------------------------------
__global__ __launch_bounds__(32) void gateUpKernel(const _Float16* __restrict__ A,
                                                   const _Float16* __restrict__ Bg,
                                                   const _Float16* __restrict__ Bu,
                                                   _Float16* __restrict__ H,
                                                   int M, int N, int K) {
  __shared__ __align__(16) _Float16 Alds[2][TILE_ELE];
  __shared__ __align__(16) _Float16 Bgl[2][32 * LPITCH + 16];
  __shared__ __align__(16) _Float16 Bul[2][32 * LPITCH + 16];
  int lane = threadIdx.x, half = lane >> 4, ln = lane & 15;
  int tn = blockIdx.x * 32, tm = blockIdx.y * 64;
  const _Float16* Ag  = A  + (size_t)tm * K;
  const _Float16* Bgg = Bg + (size_t)tn * K;
  const _Float16* Bug = Bu + (size_t)tn * K;

  v8f ag[4][2], au[4][2];
#pragma unroll
  for (int ms = 0; ms < 4; ++ms)
#pragma unroll
    for (int ns = 0; ns < 2; ++ns) { ag[ms][ns] = (v8f){}; au[ms][ns] = (v8f){}; }

  const int nsteps = K / 64;
#if HAS_TDM
  tdm_load_2d(Ag,  lds_addr_of(&Alds[0][0]), 64, K);
  tdm_load_2d(Bgg, lds_addr_of(&Bgl[0][0]),  32, K);
  tdm_load_2d(Bug, lds_addr_of(&Bul[0][0]),  32, K);
  for (int s = 0; s < nsteps; ++s) {
    int cur = s & 1;
    if (s + 1 < nsteps) {
      tdm_load_2d(Ag  + (s + 1) * 64, lds_addr_of(&Alds[cur ^ 1][0]), 64, K);
      tdm_load_2d(Bgg + (s + 1) * 64, lds_addr_of(&Bgl[cur ^ 1][0]),  32, K);
      tdm_load_2d(Bug + (s + 1) * 64, lds_addr_of(&Bul[cur ^ 1][0]),  32, K);
      __builtin_amdgcn_s_wait_tensorcnt(3);
    } else {
      __builtin_amdgcn_s_wait_tensorcnt(0);
    }
    __syncthreads();
#else
  for (int s = 0; s < nsteps; ++s) {
    int cur = 0;
    copy_tile(Ag  + s * 64, &Alds[0][0], 64, K);
    copy_tile(Bgg + s * 64, &Bgl[0][0],  32, K);
    copy_tile(Bug + s * 64, &Bul[0][0],  32, K);
    __syncthreads();
#endif
#pragma unroll
    for (int kk = 0; kk < 64; kk += 32) {
      v16h af[4], bg2[2], bu2[2];
#pragma unroll
      for (int ms = 0; ms < 4; ++ms)
        af[ms] = load_a_frag(&Alds[cur][0] + (ms * 16 + ln) * LPITCH + kk, half);
#pragma unroll
      for (int ns = 0; ns < 2; ++ns) {
        bg2[ns] = *(const v16h*)(&Bgl[cur][0] + (ns * 16 + ln) * LPITCH + kk + half * 16);
        bu2[ns] = *(const v16h*)(&Bul[cur][0] + (ns * 16 + ln) * LPITCH + kk + half * 16);
      }
#pragma unroll
      for (int ms = 0; ms < 4; ++ms)
#pragma unroll
        for (int ns = 0; ns < 2; ++ns) {
          ag[ms][ns] = wmma_f16(af[ms], bg2[ns], ag[ms][ns]);
          au[ms][ns] = wmma_f16(af[ms], bu2[ns], au[ms][ns]);
        }
    }
    __syncthreads();
  }

#pragma unroll
  for (int ms = 0; ms < 4; ++ms)
#pragma unroll
    for (int ns = 0; ns < 2; ++ns)
#pragma unroll
      for (int i = 0; i < 8; ++i) {
        float g = ag[ms][ns][i];
        float u = au[ms][ns][i];
        float silu = g / (1.0f + __expf(-g));
        H[(size_t)(tm + ms * 16 + i + 8 * half) * N + tn + ns * 16 + ln] =
            (_Float16)(silu * u);
      }
}

// ---------------------------------------------------------------------------
// Flash attention: one wave per (batch, head, 16 q-rows). GQA 16->4.
// ---------------------------------------------------------------------------
__global__ __launch_bounds__(32) void attnKernel(const _Float16* __restrict__ Qh,
                                                 const _Float16* __restrict__ Kh,
                                                 const _Float16* __restrict__ Vt,
                                                 _Float16* __restrict__ attn_h) {
  const int qt = blockIdx.x;
  const int h  = blockIdx.y;
  const int b  = blockIdx.z;
  const int hkv = h >> 2;
  const int lane = threadIdx.x;
  const int half = lane >> 4;
  const int ln   = lane & 15;
  const int qbase = qt * 16;
  const float scale = 0.125f;

  const _Float16* Qrow = Qh + ((size_t)(b * SEQ) + qbase + ln) * D_MODEL + h * HEAD_DIM;

  __shared__ __align__(32) _Float16 Plds[16 * 32];

  float m_r[8], l_r[8];
  v8f o[4];
#pragma unroll
  for (int i = 0; i < 8; ++i) { m_r[i] = -1e30f; l_r[i] = 0.f; }
#pragma unroll
  for (int t = 0; t < 4; ++t) o[t] = (v8f){};

  const int kmax = qbase + 15;
  for (int k0 = 0; k0 <= kmax; k0 += 32) {
    if (k0 + 32 <= kmax) {  // prefetch next kv block's K rows (global_prefetch_b8)
      __builtin_prefetch(Kh + ((size_t)(b * SEQ) + k0 + 32 + ln) * KV_DIM
                            + hkv * HEAD_DIM, 0, 0);
    }
    v8f sa = {}, sb = {};
    for (int d0 = 0; d0 < HEAD_DIM; d0 += 32) {
      v16h aq = load_a_frag(Qrow + d0, half);
      const _Float16* Kp = Kh + ((size_t)(b * SEQ) + k0 + ln) * KV_DIM
                              + hkv * HEAD_DIM + d0 + half * 16;
      v16h bk0 = *(const v16h*)Kp;
      v16h bk1 = *(const v16h*)(Kp + (size_t)16 * KV_DIM);
      sa = wmma_f16(aq, bk0, sa);
      sb = wmma_f16(aq, bk1, sb);
    }

#pragma unroll
    for (int i = 0; i < 8; ++i) {
      int qrow = qbase + i + 8 * half;
      float va = sa[i] * scale;
      float vb = sb[i] * scale;
      if (k0 + ln > qrow)      va = -1e30f;
      if (k0 + 16 + ln > qrow) vb = -1e30f;
      float mx = fmaxf(va, vb);
#pragma unroll
      for (int off = 1; off < 16; off <<= 1)
        mx = fmaxf(mx, __shfl_xor(mx, off, 32));
      float mnew  = fmaxf(m_r[i], mx);
      float alpha = __expf(m_r[i] - mnew);
      float pa = __expf(va - mnew);
      float pb = __expf(vb - mnew);
      float ps = pa + pb;
#pragma unroll
      for (int off = 1; off < 16; off <<= 1)
        ps += __shfl_xor(ps, off, 32);
      l_r[i] = l_r[i] * alpha + ps;
      m_r[i] = mnew;
#pragma unroll
      for (int t = 0; t < 4; ++t) o[t][i] *= alpha;
      Plds[(i + 8 * half) * 32 + ln]      = (_Float16)pa;
      Plds[(i + 8 * half) * 32 + 16 + ln] = (_Float16)pb;
    }
    __syncthreads();

    v16h ap = load_a_frag(Plds + ln * 32, half);
#pragma unroll
    for (int t = 0; t < 4; ++t) {
      const _Float16* Vp = Vt + (size_t)(hkv * HEAD_DIM + t * 16 + ln) * TOKENS
                              + b * SEQ + k0 + half * 16;
      v16h bv = *(const v16h*)Vp;
      o[t] = wmma_f16(ap, bv, o[t]);
    }
    __syncthreads();
  }

#pragma unroll
  for (int t = 0; t < 4; ++t) {
#pragma unroll
    for (int i = 0; i < 8; ++i) {
      int q = qbase + i + 8 * half;
      int d = t * 16 + ln;
      attn_h[((size_t)(b * SEQ) + q) * D_MODEL + h * HEAD_DIM + d] =
          (_Float16)(o[t][i] / l_r[i]);
    }
  }
}

// ---------------------------------------------------------------------------
// out = x + attn_proj + ffn_out
// ---------------------------------------------------------------------------
__global__ void add3Kernel(const float* __restrict__ x,
                           const float* __restrict__ a,
                           const float* __restrict__ f,
                           float* __restrict__ out, long n) {
  long idx = (long)blockIdx.x * 256 + threadIdx.x;
  if (idx < n) out[idx] = x[idx] + a[idx] + f[idx];
}

// ---------------------------------------------------------------------------
extern "C" void kernel_launch(void* const* d_in, const int* in_sizes, int n_in,
                              void* d_out, int out_size, void* d_ws, size_t ws_size,
                              hipStream_t stream) {
  const float* x    = (const float*)d_in[0];
  const float* ln_w = (const float*)d_in[1];
  const float* ln_b = (const float*)d_in[2];
  const float* wq   = (const float*)d_in[3];
  const float* wk   = (const float*)d_in[4];
  const float* wv   = (const float*)d_in[5];
  const float* wo   = (const float*)d_in[6];
  const float* wg   = (const float*)d_in[7];
  const float* wu   = (const float*)d_in[8];
  const float* wd   = (const float*)d_in[9];
  float* out = (float*)d_out;

  char* p = (char*)d_ws;
  auto take = [&](size_t bytes) -> char* {
    char* q = p;
    p += (bytes + 255) & ~(size_t)255;
    return q;
  };
  _Float16* normed = (_Float16*)take((size_t)TOKENS * D_MODEL * 2);
  _Float16* wq_t   = (_Float16*)take((size_t)D_MODEL * D_MODEL * 2);
  _Float16* wk_t   = (_Float16*)take((size_t)KV_DIM * D_MODEL * 2);
  _Float16* wv_t   = (_Float16*)take((size_t)KV_DIM * D_MODEL * 2);
  _Float16* wo_t   = (_Float16*)take((size_t)D_MODEL * D_MODEL * 2);
  _Float16* wg_t   = (_Float16*)take((size_t)D_FF * D_MODEL * 2);
  _Float16* wu_t   = (_Float16*)take((size_t)D_FF * D_MODEL * 2);
  _Float16* wd_t   = (_Float16*)take((size_t)D_MODEL * D_FF * 2);
  _Float16* Q_h    = (_Float16*)take((size_t)TOKENS * D_MODEL * 2);
  _Float16* K_h    = (_Float16*)take((size_t)TOKENS * KV_DIM * 2);
  _Float16* V_t    = (_Float16*)take((size_t)KV_DIM * TOKENS * 2);
  _Float16* attn_h = (_Float16*)take((size_t)TOKENS * D_MODEL * 2);
  float*    Aout   = (float*)take((size_t)TOKENS * D_MODEL * 4);
  _Float16* Hbuf   = (_Float16*)take((size_t)TOKENS * D_FF * 2);
  float*    Fout   = (float*)take((size_t)TOKENS * D_MODEL * 4);

  auto ct = [&](const float* W, _Float16* Wt, int K, int N) {
    long n = (long)K * N;
    castTransposeKernel<<<(unsigned)((n + 255) / 256), 256, 0, stream>>>(W, Wt, K, N);
  };
  ct(wq, wq_t, D_MODEL, D_MODEL);
  ct(wk, wk_t, D_MODEL, KV_DIM);
  ct(wv, wv_t, D_MODEL, KV_DIM);
  ct(wo, wo_t, D_MODEL, D_MODEL);
  ct(wg, wg_t, D_MODEL, D_FF);
  ct(wu, wu_t, D_MODEL, D_FF);
  ct(wd, wd_t, D_FF, D_MODEL);

  lnKernel<<<TOKENS, 256, 0, stream>>>(x, ln_w, ln_b, normed);

  gemmWmmaTiled<<<dim3(D_MODEL / 64, TOKENS / 64), 32, 0, stream>>>(
      normed, wq_t, (void*)Q_h, TOKENS, D_MODEL, D_MODEL, /*f16*/1, 0);
  gemmWmmaTiled<<<dim3(KV_DIM / 64, TOKENS / 64), 32, 0, stream>>>(
      normed, wk_t, (void*)K_h, TOKENS, KV_DIM, D_MODEL, /*f16*/1, 0);
  gemmWmmaTiled<<<dim3(KV_DIM / 64, TOKENS / 64), 32, 0, stream>>>(
      normed, wv_t, (void*)V_t, TOKENS, KV_DIM, D_MODEL, /*f16 transposed*/2, TOKENS);

  attnKernel<<<dim3(SEQ / 16, N_HEADS, BATCH), 32, 0, stream>>>(Q_h, K_h, V_t, attn_h);

  gemmWmmaTiled<<<dim3(D_MODEL / 64, TOKENS / 64), 32, 0, stream>>>(
      attn_h, wo_t, (void*)Aout, TOKENS, D_MODEL, D_MODEL, /*f32*/0, 0);

  gateUpKernel<<<dim3(D_FF / 32, TOKENS / 64), 32, 0, stream>>>(
      normed, wg_t, wu_t, Hbuf, TOKENS, D_FF, D_MODEL);

  gemmWmmaTiled<<<dim3(D_MODEL / 64, TOKENS / 64), 32, 0, stream>>>(
      Hbuf, wd_t, (void*)Fout, TOKENS, D_MODEL, D_FF, /*f32*/0, 0);

  long n = (long)TOKENS * D_MODEL;
  add3Kernel<<<(unsigned)((n + 255) / 256), 256, 0, stream>>>(x, Aout, Fout, out, n);
}
